// MoE_55697135894810
// MI455X (gfx1250) — matrix operand order repaired and checked by reading, not compiled, not run on hardware
//
#include <hip/hip_runtime.h>
#include <hip/hip_bf16.h>
#include <math.h>

// ---------------------------------------------------------------------------
// MoE (top-2 of 8 routed experts + 2 shared experts), bf16 WMMA pipeline.
//   x:[2,1024,1024] f32, gate_w:[8,1024], W1:[8,1536,1024], W2:[8,1024,768],
//   Ws1:[3072,1024], Ws2:[1024,1536] -> out:[2,1024,1024] f32
// Pipeline: cvt f32->bf16 | gate+top2 route | fc1+SwiGLU (routed gathered,
// shared dense) | fc2 (shared writes out, routed atomic-adds weighted).
// GEMMs: register-staged, LDS double-buffered, one barrier per K-step so the
// global-load latency of tile k+1 hides behind the WMMAs of tile k.
// ---------------------------------------------------------------------------

typedef __bf16 bf16;
typedef __attribute__((ext_vector_type(4)))  bf16  v4bf;
typedef __attribute__((ext_vector_type(8)))  bf16  v8bf;
typedef __attribute__((ext_vector_type(16))) bf16  v16bf;
typedef __attribute__((ext_vector_type(8)))  float v8f;

union V16U { v16bf v; v8bf h[2]; };

// Block = 256 threads = 8 waves; block tile 64(M) x 128(N);
// wave tile 32x32 as 2x2 WMMA 16x16 tiles; K-step 32 (= one wmma depth).
constexpr int TM = 64;
constexpr int TN = 128;
constexpr int KT = 32;
constexpr int LDS_PAD = 8;                 // 16B pad -> conflict-free frag reads
constexpr int LDS_LD = KT + LDS_PAD;       // row stride (elements)

// A-fragment (16x32 bf16): lane m<16 holds row m, K 0..7 & 16..23;
// lane m+16 holds K 8..15 & 24..31 (ISA 7.12.2).
__device__ __forceinline__ v16bf load_fragA(const bf16* As, int row, int half) {
  V16U u;
  u.h[0] = *(const v8bf*)(As + row * LDS_LD + half * 8);
  u.h[1] = *(const v8bf*)(As + row * LDS_LD + 16 + half * 8);
  return u.v;
}
// B-fragment (32x16 bf16): lane n<16 holds column n, K 0..7 & 16..23; lane n+16 K 8..15 & 24..31.
__device__ __forceinline__ v16bf load_fragB(const bf16* Bs, int col, int half) {
  V16U u;
  u.h[0] = *(const v8bf*)(Bs + col * LDS_LD + half * 8);
  u.h[1] = *(const v8bf*)(Bs + col * LDS_LD + 16 + half * 8);
  return u.v;
}

__device__ __forceinline__ float fast_silu(float g) {
  // g * sigmoid(g) with one v_rcp_f32 instead of an IEEE divide sequence
  return g * __builtin_amdgcn_rcpf(1.0f + __expf(-g));
}

// ---------------------------------------------------------------------------
// fc1 + SwiGLU.  C_y = A @ Wy^T, C_g = A @ Wg^T in one K-loop; epilogue stores
// act = y * silu(g) as bf16.  GATHER: A rows gathered via per-expert token list.
// ---------------------------------------------------------------------------
template<bool GATHER>
__global__ __launch_bounds__(256) void fc1_swiglu(
    const bf16* __restrict__ X,       // [T, Kdim]
    const bf16* __restrict__ Wall,    // [E][2*Hspan, Kdim] (E=1 if !GATHER)
    bf16* __restrict__ ActAll,        // [E][T, Hspan]
    const int* __restrict__ counts,   // [E]
    const int* __restrict__ tokList,  // [E, T]
    int Kdim, int Hspan, int T)
{
  __shared__ __align__(16) bf16 As [2][TM * LDS_LD];
  __shared__ __align__(16) bf16 Bys[2][TN * LDS_LD];
  __shared__ __align__(16) bf16 Bgs[2][TN * LDS_LD];

  const int tid  = threadIdx.x;
  const int e    = GATHER ? blockIdx.z : 0;
  const int cnt  = GATHER ? counts[e] : T;
  const int mBase = blockIdx.y * TM;
  if (mBase >= cnt) return;                     // uniform: EXEC stays all-ones
  const int nBase = blockIdx.x * TN;

  const int* tl  = GATHER ? (tokList + e * T) : nullptr;
  const bf16* Wy = Wall + (size_t)e * (2 * Hspan) * Kdim + (size_t)nBase * Kdim;
  const bf16* Wg = Wy + (size_t)Hspan * Kdim;
  bf16* Act      = ActAll + (size_t)e * T * Hspan;

  // staging: A -> 1x 16B chunk/thread, B -> 2x 16B chunks/thread per matrix
  const int aRow = tid >> 2;
  const int aOff = (tid & 3) * 8;
  int gRow;
  if (GATHER) { int r = mBase + aRow; if (r >= cnt) r = cnt - 1; gRow = tl[r]; }
  else        { gRow = mBase + aRow; }
  const bf16* aSrc = X + (size_t)gRow * Kdim + aOff;

  const int bRow = tid >> 1;
  const int bOff = (tid & 1) * 16;
  const bf16* bySrc = Wy + (size_t)bRow * Kdim + bOff;
  const bf16* bgSrc = Wg + (size_t)bRow * Kdim + bOff;

  const int wave = tid >> 5, lane = tid & 31;
  const int waveM = wave >> 2, waveN = wave & 3;
  const int half = lane >> 4, l15 = lane & 15;

  v8f accY[2][2] = {};
  v8f accG[2][2] = {};

  v8bf rA, rY0, rY1, rG0, rG1;                  // register staging for tile k+1
  auto gload = [&](int k) {
    rA  = *(const v8bf*)(aSrc + k);
    rY0 = *(const v8bf*)(bySrc + k);
    rY1 = *(const v8bf*)(bySrc + k + 8);
    rG0 = *(const v8bf*)(bgSrc + k);
    rG1 = *(const v8bf*)(bgSrc + k + 8);
  };
  auto lstore = [&](int b) {
    *(v8bf*)(As[b]  + aRow * LDS_LD + aOff)     = rA;
    *(v8bf*)(Bys[b] + bRow * LDS_LD + bOff)     = rY0;
    *(v8bf*)(Bys[b] + bRow * LDS_LD + bOff + 8) = rY1;
    *(v8bf*)(Bgs[b] + bRow * LDS_LD + bOff)     = rG0;
    *(v8bf*)(Bgs[b] + bRow * LDS_LD + bOff + 8) = rG1;
  };

  gload(0);
  lstore(0);
  __syncthreads();

  const int nk = Kdim / KT;
  for (int i = 0; i < nk; ++i) {
    const int b = i & 1;
    if (i + 1 < nk) gload((i + 1) * KT);        // issue next-tile global loads

    v16bf a[2], by[2], bg[2];
    #pragma unroll
    for (int mi = 0; mi < 2; ++mi)
      a[mi] = load_fragA(As[b], waveM * 32 + mi * 16 + l15, half);
    #pragma unroll
    for (int ni = 0; ni < 2; ++ni) {
      by[ni] = load_fragB(Bys[b], waveN * 32 + ni * 16 + l15, half);
      bg[ni] = load_fragB(Bgs[b], waveN * 32 + ni * 16 + l15, half);
    }
    #pragma unroll
    for (int mi = 0; mi < 2; ++mi)
      #pragma unroll
      for (int ni = 0; ni < 2; ++ni) {
        accY[mi][ni] = __builtin_amdgcn_wmma_f32_16x16x32_bf16(
            false, a[mi], false, by[ni], (short)0, accY[mi][ni], false, false);
        accG[mi][ni] = __builtin_amdgcn_wmma_f32_16x16x32_bf16(
            false, a[mi], false, bg[ni], (short)0, accG[mi][ni], false, false);
      }

    if (i + 1 < nk) lstore(b ^ 1);              // loadcnt wait hides behind wmma
    __syncthreads();
  }

  // C/D layout (ISA 7.12.2): lane l -> col = l&15, row = (l>>4)*8 + vgpr
  #pragma unroll
  for (int mi = 0; mi < 2; ++mi)
    #pragma unroll
    for (int ni = 0; ni < 2; ++ni) {
      const int col = nBase + waveN * 32 + ni * 16 + l15;
      #pragma unroll
      for (int v = 0; v < 8; ++v) {
        const int row = mBase + waveM * 32 + mi * 16 + half * 8 + v;
        float y = accY[mi][ni][v];
        float g = accG[mi][ni][v];
        Act[(size_t)row * Hspan + col] = (bf16)(y * fast_silu(g));
      }
    }
}

// ---------------------------------------------------------------------------
// fc2.  C = Act @ W^T.  ROUTED: scale by gate weight, atomic scatter-add to
// out[token]; else plain store (shared experts, runs first in-stream).
// ---------------------------------------------------------------------------
template<bool ROUTED>
__global__ __launch_bounds__(256) void fc2_combine(
    const bf16* __restrict__ ActAll,  // [E][T, Kdim]
    const bf16* __restrict__ Wall,    // [E][D, Kdim]
    float* __restrict__ Out,          // [T, D]
    const int* __restrict__ counts,
    const int* __restrict__ tokList,
    const float* __restrict__ wtList,
    int Kdim, int T, int D)
{
  __shared__ __align__(16) bf16 As[2][TM * LDS_LD];
  __shared__ __align__(16) bf16 Bs[2][TN * LDS_LD];

  const int tid = threadIdx.x;
  const int e   = ROUTED ? blockIdx.z : 0;
  const int cnt = ROUTED ? counts[e] : T;
  const int mBase = blockIdx.y * TM;
  if (mBase >= cnt) return;
  const int nBase = blockIdx.x * TN;

  const bf16* A = ActAll + (size_t)e * T * Kdim;
  const bf16* W = Wall + (size_t)e * D * Kdim + (size_t)nBase * Kdim;

  const int aRow = tid >> 2;
  const int aOff = (tid & 3) * 8;
  const bf16* aSrc = A + (size_t)(mBase + aRow) * Kdim + aOff;

  const int bRow = tid >> 1;
  const int bOff = (tid & 1) * 16;
  const bf16* bSrc = W + (size_t)bRow * Kdim + bOff;

  const int wave = tid >> 5, lane = tid & 31;
  const int waveM = wave >> 2, waveN = wave & 3;
  const int half = lane >> 4, l15 = lane & 15;

  v8f acc[2][2] = {};

  v8bf rA, rB0, rB1;
  auto gload = [&](int k) {
    rA  = *(const v8bf*)(aSrc + k);
    rB0 = *(const v8bf*)(bSrc + k);
    rB1 = *(const v8bf*)(bSrc + k + 8);
  };
  auto lstore = [&](int b) {
    *(v8bf*)(As[b] + aRow * LDS_LD + aOff)     = rA;
    *(v8bf*)(Bs[b] + bRow * LDS_LD + bOff)     = rB0;
    *(v8bf*)(Bs[b] + bRow * LDS_LD + bOff + 8) = rB1;
  };

  gload(0);
  lstore(0);
  __syncthreads();

  const int nk = Kdim / KT;
  for (int i = 0; i < nk; ++i) {
    const int b = i & 1;
    if (i + 1 < nk) gload((i + 1) * KT);

    v16bf a[2], bb[2];
    #pragma unroll
    for (int mi = 0; mi < 2; ++mi)
      a[mi] = load_fragA(As[b], waveM * 32 + mi * 16 + l15, half);
    #pragma unroll
    for (int ni = 0; ni < 2; ++ni)
      bb[ni] = load_fragB(Bs[b], waveN * 32 + ni * 16 + l15, half);
    #pragma unroll
    for (int mi = 0; mi < 2; ++mi)
      #pragma unroll
      for (int ni = 0; ni < 2; ++ni)
        acc[mi][ni] = __builtin_amdgcn_wmma_f32_16x16x32_bf16(
            false, a[mi], false, bb[ni], (short)0, acc[mi][ni], false, false);

    if (i + 1 < nk) lstore(b ^ 1);
    __syncthreads();
  }

  const int*   tl = ROUTED ? (tokList + e * T) : nullptr;
  const float* wl = ROUTED ? (wtList + e * T) : nullptr;
  #pragma unroll
  for (int mi = 0; mi < 2; ++mi)
    #pragma unroll
    for (int ni = 0; ni < 2; ++ni) {
      const int col = nBase + waveN * 32 + ni * 16 + l15;
      #pragma unroll
      for (int v = 0; v < 8; ++v) {
        const int row = mBase + waveM * 32 + mi * 16 + half * 8 + v;
        float val = acc[mi][ni][v];
        if (ROUTED) {
          if (row < cnt)
            atomicAdd(Out + (size_t)tl[row] * D + col, val * wl[row]);
        } else {
          Out[(size_t)row * D + col] = val;
        }
      }
    }
}

// ---------------------------------------------------------------------------
// Gate: scores = softmax(x @ gate_w^T), top-2; scatter (token, weight) into
// per-expert lists.  One wave per token; gate_w staged in LDS (32KB).
// ---------------------------------------------------------------------------
__global__ __launch_bounds__(256) void gate_topk(
    const float* __restrict__ X, const float* __restrict__ Gw,
    int* __restrict__ counts, int* __restrict__ tokList,
    float* __restrict__ wtList, int T, int D)
{
  __shared__ float gws[8 * 1024];
  const int tid = threadIdx.x;
  for (int i = tid; i < 8 * D; i += 256) gws[i] = Gw[i];
  __syncthreads();

  const int wave = tid >> 5, lane = tid & 31;
  const int t = blockIdx.x * 8 + wave;
  if (t >= T) return;

  const float* xr = X + (size_t)t * D;
  float acc[8] = {0.f, 0.f, 0.f, 0.f, 0.f, 0.f, 0.f, 0.f};
  for (int d = lane; d < D; d += 32) {
    float xv = xr[d];
    #pragma unroll
    for (int e = 0; e < 8; ++e) acc[e] = fmaf(xv, gws[e * D + d], acc[e]);
  }
  #pragma unroll
  for (int e = 0; e < 8; ++e)
    for (int o = 16; o > 0; o >>= 1) acc[e] += __shfl_xor(acc[e], o, 32);

  if (lane == 0) {
    float m = acc[0];
    #pragma unroll
    for (int e = 1; e < 8; ++e) m = fmaxf(m, acc[e]);
    float p[8], s = 0.f;
    #pragma unroll
    for (int e = 0; e < 8; ++e) { p[e] = __expf(acc[e] - m); s += p[e]; }
    float inv = 1.0f / s;    // one scalar IEEE divide per token: keep exact
    int i1 = 0; float b1 = p[0];
    #pragma unroll
    for (int e = 1; e < 8; ++e) if (p[e] > b1) { b1 = p[e]; i1 = e; }
    int i2 = (i1 == 0) ? 1 : 0; float b2 = p[i2];
    #pragma unroll
    for (int e = 0; e < 8; ++e)
      if (e != i1 && p[e] > b2) { b2 = p[e]; i2 = e; }
    int pos = atomicAdd(&counts[i1], 1);
    tokList[i1 * T + pos] = t; wtList[i1 * T + pos] = b1 * inv;
    pos = atomicAdd(&counts[i2], 1);
    tokList[i2 * T + pos] = t; wtList[i2 * T + pos] = b2 * inv;
  }
}

// fp32 -> bf16 bulk conversion (amortized: weight tiles are re-read ~32x).
__global__ __launch_bounds__(256) void cvt_f32_to_bf16(
    const float* __restrict__ s, bf16* __restrict__ d, int n)
{
  int i = (blockIdx.x * 256 + threadIdx.x) * 4;
  if (i + 3 < n) {
    float4 f = *(const float4*)(s + i);
    v4bf o;
    o[0] = (bf16)f.x; o[1] = (bf16)f.y; o[2] = (bf16)f.z; o[3] = (bf16)f.w;
    *(v4bf*)(d + i) = o;
  }
}

// ---------------------------------------------------------------------------
extern "C" void kernel_launch(void* const* d_in, const int* in_sizes, int n_in,
                              void* d_out, int out_size, void* d_ws, size_t ws_size,
                              hipStream_t stream)
{
  const float* x   = (const float*)d_in[0];
  const float* gw  = (const float*)d_in[1];
  const float* W1  = (const float*)d_in[2];
  const float* W2  = (const float*)d_in[3];
  const float* Ws1 = (const float*)d_in[4];
  const float* Ws2 = (const float*)d_in[5];
  float* out = (float*)d_out;

  const int D = 1024, H = 768, E = 8, NSH = 1536;   // NSH = NS*H
  const int T = in_sizes[0] / D;                     // 2048 tokens

  char* wsb = (char*)d_ws; size_t off = 0;
  auto alloc = [&](size_t bytes) -> void* {
    void* p = wsb + off; off += (bytes + 255) & ~(size_t)255; return p;
  };
  bf16*  xb   = (bf16*) alloc((size_t)T * D * 2);
  bf16*  w1b  = (bf16*) alloc((size_t)E * 2 * H * D * 2);
  bf16*  w2b  = (bf16*) alloc((size_t)E * D * H * 2);
  bf16*  ws1b = (bf16*) alloc((size_t)2 * NSH * D * 2);
  bf16*  ws2b = (bf16*) alloc((size_t)D * NSH * 2);
  bf16*  actR = (bf16*) alloc((size_t)E * T * H * 2);
  bf16*  actS = (bf16*) alloc((size_t)T * NSH * 2);
  int*   cnts = (int*)  alloc((size_t)E * 4);
  int*   tokL = (int*)  alloc((size_t)E * T * 4);
  float* wtL  = (float*)alloc((size_t)E * T * 4);
  (void)ws_size; (void)n_in; (void)out_size;

  hipMemsetAsync(cnts, 0, E * sizeof(int), stream);

  auto cvt = [&](const float* s, bf16* d, size_t n) {
    cvt_f32_to_bf16<<<(int)(n / 1024), 256, 0, stream>>>(s, d, (int)n);
  };
  cvt(x,   xb,   (size_t)T * D);
  cvt(W1,  w1b,  (size_t)E * 2 * H * D);
  cvt(W2,  w2b,  (size_t)E * D * H);
  cvt(Ws1, ws1b, (size_t)2 * NSH * D);
  cvt(Ws2, ws2b, (size_t)D * NSH);

  gate_topk<<<T / 8, 256, 0, stream>>>(x, gw, cnts, tokL, wtL, T, D);

  // routed fc1 (+SwiGLU), gathered per expert; worst-case grid, early exit
  fc1_swiglu<true><<<dim3(H / TN, T / TM, E), 256, 0, stream>>>(
      xb, w1b, actR, cnts, tokL, D, H, T);
  // shared fc1 (+SwiGLU), dense
  fc1_swiglu<false><<<dim3(NSH / TN, T / TM, 1), 256, 0, stream>>>(
      xb, ws1b, actS, nullptr, nullptr, D, NSH, T);
  // shared fc2 writes out (must precede routed adds; in-stream order)
  fc2_combine<false><<<dim3(D / TN, T / TM, 1), 256, 0, stream>>>(
      actS, ws2b, out, nullptr, nullptr, nullptr, NSH, T, D);
  // routed fc2: weighted atomic scatter-add
  fc2_combine<true><<<dim3(D / TN, T / TM, E), 256, 0, stream>>>(
      actR, w2b, out, cnts, tokL, wtL, H, T, D);
}
